// HybridPatchEmbed_28561532518656
// MI455X (gfx1250) — compile-verified, matrix operand order checked by reading, not run on hardware
//
#include <hip/hip_runtime.h>
#include <hip/hip_bf16.h>

typedef _Float16 v8h  __attribute__((ext_vector_type(8)));
typedef _Float16 v16h __attribute__((ext_vector_type(16)));
typedef float    v8f  __attribute__((ext_vector_type(8)));

#define EPSV 1e-5f

// ---------------------------------------------------------------------------
// CDNA5 async global->LDS copy (ASYNCcnt path), 16B per lane per issue.
// lds_off: LDS byte offset (low 32 bits of generic shared pointer).
// ---------------------------------------------------------------------------
__device__ __forceinline__ void async_copy_b128(uint32_t lds_off, const void* gptr) {
  asm volatile("global_load_async_to_lds_b128 %0, %1, off"
               :: "v"(lds_off), "v"((unsigned long long)(uintptr_t)gptr)
               : "memory");
}
__device__ __forceinline__ void wait_async_le1() {
  asm volatile("s_wait_asynccnt 0x1" ::: "memory");
}
__device__ __forceinline__ void wait_async_le0() {
  asm volatile("s_wait_asynccnt 0x0" ::: "memory");
}

// ---------------------------------------------------------------------------
// Kernel 1: transpose f32 (K x N) -> f16 (N x K)
// ---------------------------------------------------------------------------
__global__ __launch_bounds__(256) void transpose_half_kernel(
    const float* __restrict__ src, _Float16* __restrict__ dst, int K, int N) {
  int idx = blockIdx.x * 256 + threadIdx.x;
  if (idx < K * N) {
    int k = idx / N, n = idx - k * N;
    dst[(size_t)n * K + k] = (_Float16)src[idx];
  }
}

// ---------------------------------------------------------------------------
// Kernel 2: fused front-end. One block per batch element, 256 threads.
// ---------------------------------------------------------------------------
__global__ __launch_bounds__(256) void tokens_kernel(
    const float* __restrict__ bitboards,
    const float* __restrict__ conv1_w, const float* __restrict__ conv1_b,
    const float* __restrict__ bn1_g, const float* __restrict__ bn1_b,
    const float* __restrict__ bn1_m, const float* __restrict__ bn1_v,
    const float* __restrict__ conv2_w, const float* __restrict__ conv2_b,
    const float* __restrict__ bn2_g, const float* __restrict__ bn2_b,
    const float* __restrict__ bn2_m, const float* __restrict__ bn2_v,
    const float* __restrict__ proj_w, const float* __restrict__ proj_b,
    const float* __restrict__ sq_w, const float* __restrict__ sq_b,
    const float* __restrict__ sq_ln_g, const float* __restrict__ sq_ln_b,
    const float* __restrict__ init_w, const float* __restrict__ init_b,
    float* __restrict__ tokens_f32, _Float16* __restrict__ tokens_h) {
  const int b = blockIdx.x;
  const int t = threadIdx.x;

  __shared__ float bitb[14][64];
  __shared__ float h1[32][64];
  __shared__ float comb[48][64];

  for (int i = t; i < 14 * 64; i += 256)
    bitb[i >> 6][i & 63] = bitboards[(size_t)b * 14 * 64 + i];
  __syncthreads();

  // conv1 (3x3, pad1) + bn1 + relu
  for (int i = t; i < 32 * 64; i += 256) {
    int h = i >> 6, s = i & 63, y = s >> 3, x = s & 7;
    float acc = conv1_b[h];
    for (int c = 0; c < 14; ++c)
      for (int ky = 0; ky < 3; ++ky) {
        int yy = y + ky - 1;
        if ((unsigned)yy > 7u) continue;
        for (int kx = 0; kx < 3; ++kx) {
          int xx = x + kx - 1;
          if ((unsigned)xx > 7u) continue;
          acc += bitb[c][yy * 8 + xx] * conv1_w[((h * 14 + c) * 3 + ky) * 3 + kx];
        }
      }
    float sc = bn1_g[h] * rsqrtf(bn1_v[h] + EPSV);
    acc = acc * sc + (bn1_b[h] - bn1_m[h] * sc);
    h1[h][s] = fmaxf(acc, 0.f);
  }
  __syncthreads();

  // conv2 + bn2, + 1x1 proj residual, relu -> comb[0..31]
  for (int i = t; i < 32 * 64; i += 256) {
    int h = i >> 6, s = i & 63, y = s >> 3, x = s & 7;
    float acc = conv2_b[h];
    for (int c = 0; c < 32; ++c)
      for (int ky = 0; ky < 3; ++ky) {
        int yy = y + ky - 1;
        if ((unsigned)yy > 7u) continue;
        for (int kx = 0; kx < 3; ++kx) {
          int xx = x + kx - 1;
          if ((unsigned)xx > 7u) continue;
          acc += h1[c][yy * 8 + xx] * conv2_w[((h * 32 + c) * 3 + ky) * 3 + kx];
        }
      }
    float sc = bn2_g[h] * rsqrtf(bn2_v[h] + EPSV);
    acc = acc * sc + (bn2_b[h] - bn2_m[h] * sc);
    float pr = proj_b[h];
    for (int c = 0; c < 14; ++c) pr += bitb[c][s] * proj_w[h * 14 + c];
    comb[h][s] = fmaxf(acc + pr, 0.f);
  }
  __syncthreads();

  // per-square embedding: 13 -> 16, LN over 16, relu -> comb[32..47]
  if (t < 64) {
    int s = t;
    float oh[13], psum = 0.f;
    for (int c = 0; c < 12; ++c) { oh[c] = bitb[c][s]; psum += oh[c]; }
    oh[12] = 1.f - fminf(fmaxf(psum, 0.f), 1.f);
    float gv[16], mu = 0.f;
    for (int p = 0; p < 16; ++p) {
      float a = sq_b[s * 16 + p];
      for (int c = 0; c < 13; ++c) a += oh[c] * sq_w[(s * 13 + c) * 16 + p];
      gv[p] = a;
      mu += a;
    }
    mu *= (1.f / 16.f);
    float var = 0.f;
    for (int p = 0; p < 16; ++p) { float d = gv[p] - mu; var += d * d; }
    var *= (1.f / 16.f);
    float inv = rsqrtf(var + EPSV);
    for (int p = 0; p < 16; ++p)
      comb[32 + p][s] =
          fmaxf((gv[p] - mu) * inv * sq_ln_g[s * 16 + p] + sq_ln_b[s * 16 + p], 0.f);
  }
  __syncthreads();

  // 1x1 init conv: 48 -> 768; token row = b*64 + s
  for (int i = t; i < 768 * 64; i += 256) {
    int d = i >> 6, s = i & 63;
    float acc = init_b[d];
    #pragma unroll 8
    for (int c = 0; c < 48; ++c) acc += comb[c][s] * init_w[d * 48 + c];
    size_t row = (size_t)b * 64 + s;
    tokens_f32[row * 768 + d] = acc;
    tokens_h[row * 768 + d] = (_Float16)acc;
  }
}

// ---------------------------------------------------------------------------
// Kernel 3/5: f16 WMMA GEMM.  C[M x N] f32 = A[M x K] f16 * Bt[N x K]^T + bias
// Block tile 256x64, K-step 32, 256 threads (8 waves). Wave w owns two
// 16-row M-tiles x four 16-col N-tiles -> 8 v_wmma_f32_16x16x32_f16/K-step.
// A (wave-private rows) is loaded straight from global into fragments
// (2x global_load_b128 per M-tile); only B (block-shared) is staged through a
// double-buffered LDS ping-pong via global_load_async_to_lds_b128 (ASYNCcnt),
// 1 async issue per thread per K-tile.
// ---------------------------------------------------------------------------
__global__ __launch_bounds__(256) void gemm_wmma(
    const _Float16* __restrict__ A, const _Float16* __restrict__ Bt,
    const float* __restrict__ bias, float* __restrict__ C,
    int M, int N, int K) {
  constexpr int BM = 256, BN = 64, BK = 32, LDK = BK + 8;  // pad keeps 16B align
  __shared__ _Float16 Bs[2][BN][LDK];

  const int t = threadIdx.x;
  const int wave = t >> 5, lane = t & 31;
  const int lr = lane & 15, hi = lane >> 4;
  const int m0 = blockIdx.x * BM, n0 = blockIdx.y * BN;

  const uint32_t ldsB = (uint32_t)(uintptr_t)&Bs[0][0][0];
  const uint32_t bufB = (uint32_t)sizeof(_Float16) * BN * LDK;

  const v8f zero = {0.f, 0.f, 0.f, 0.f, 0.f, 0.f, 0.f, 0.f};
  v8f acc[2][4] = {{zero, zero, zero, zero}, {zero, zero, zero, zero}};

  // B staging: row = t>>2 (of 64), chunk = t&3 (16B each) -> 1 issue/thread
  const int bn = t >> 2, bq = t & 3;
  const _Float16* gB = Bt + (size_t)(n0 + bn) * K + bq * 8;
  const uint32_t lbRow = ldsB + (uint32_t)bn * (LDK * 2) + (uint32_t)bq * 16;

  // A fragment base pointers (ISA 16-bit A layout: lane<16 holds K{0..7,16..23}
  // of row base+lr, lane>=16 holds K{8..15,24..31})
  const _Float16* aPtr[2] = {
      A + (size_t)(m0 + wave * 32 + lr) * K + hi * 8,
      A + (size_t)(m0 + wave * 32 + 16 + lr) * K + hi * 8};

  const int nk = K / BK;
  async_copy_b128(lbRow, gB);  // stage B tile 0 into buf 0

  for (int i = 0; i < nk; ++i) {
    const int k0 = i * BK;
    const bool has_next = (i + 1) < nk;
    if (has_next) async_copy_b128(lbRow + (uint32_t)((i + 1) & 1) * bufB,
                                  gB + k0 + BK);
    // retire current B tile's async op (next tile's stays in flight)
    if (has_next) wait_async_le1(); else wait_async_le0();
    __syncthreads();

    const int cb = i & 1;
    #pragma unroll
    for (int mt = 0; mt < 2; ++mt) {
      // A fragment straight from global (wave-private rows; no LDS round-trip)
      v8h alo = *(const v8h*)(aPtr[mt] + k0);
      v8h ahi = *(const v8h*)(aPtr[mt] + k0 + 16);
      v16h a = __builtin_shufflevector(alo, ahi, 0, 1, 2, 3, 4, 5, 6, 7, 8, 9,
                                       10, 11, 12, 13, 14, 15);
      #pragma unroll
      for (int nt = 0; nt < 4; ++nt) {
        v8h blo = *(const v8h*)&Bs[cb][nt * 16 + lr][hi * 8];
        v8h bhi = *(const v8h*)&Bs[cb][nt * 16 + lr][16 + hi * 8];
        v16h bb = __builtin_shufflevector(blo, bhi, 0, 1, 2, 3, 4, 5, 6, 7, 8,
                                          9, 10, 11, 12, 13, 14, 15);
        acc[mt][nt] = __builtin_amdgcn_wmma_f32_16x16x32_f16(
            false, a, false, bb, (short)0, acc[mt][nt], false, false);
      }
    }
    __syncthreads();  // everyone done reading buf cb -> safe to refill next iter
  }

  // C layout: VGPR r -> M = r (lanes 0-15) / 8+r (lanes 16-31), N = lr
  #pragma unroll
  for (int mt = 0; mt < 2; ++mt) {
    #pragma unroll
    for (int nt = 0; nt < 4; ++nt) {
      int col = n0 + nt * 16 + lr;
      float bv = bias[col];
      #pragma unroll
      for (int r = 0; r < 8; ++r) {
        int row = m0 + wave * 32 + mt * 16 + hi * 8 + r;
        C[(size_t)row * N + col] = acc[mt][nt][r] + bv;
      }
    }
  }
}

// ---------------------------------------------------------------------------
// Kernel 4: row LayerNorm + exact GELU, f32 in -> f16 out. One block per row.
// ---------------------------------------------------------------------------
__global__ __launch_bounds__(256) void ln_gelu_kernel(
    const float* __restrict__ Y, const float* __restrict__ g,
    const float* __restrict__ b, _Float16* __restrict__ Out, int N) {
  int row = blockIdx.x, t = threadIdx.x;
  const float* y = Y + (size_t)row * N;
  float s = 0.f, s2 = 0.f;
  for (int i = t; i < N; i += 256) { float v = y[i]; s += v; s2 += v * v; }
  __shared__ float rs[256], rq[256];
  rs[t] = s; rq[t] = s2;
  __syncthreads();
  for (int o = 128; o > 0; o >>= 1) {
    if (t < o) { rs[t] += rs[t + o]; rq[t] += rq[t + o]; }
    __syncthreads();
  }
  float mu = rs[0] / N;
  float inv = rsqrtf(rq[0] / N - mu * mu + EPSV);
  for (int i = t; i < N; i += 256) {
    float v = (y[i] - mu) * inv * g[i] + b[i];
    float ge = 0.5f * v * (1.f + erff(v * 0.70710678118654752f));
    Out[(size_t)row * N + i] = (_Float16)ge;
  }
}

// ---------------------------------------------------------------------------
// Kernel 6: row LayerNorm (in-place on d_out, bias already fused by GEMM)
// + residual add of tokens. One block per row.
// ---------------------------------------------------------------------------
__global__ __launch_bounds__(256) void ln_residual_kernel(
    float* __restrict__ Y, const float* __restrict__ tokens,
    const float* __restrict__ g, const float* __restrict__ b, int N) {
  int row = blockIdx.x, t = threadIdx.x;
  float* y = Y + (size_t)row * N;
  const float* tok = tokens + (size_t)row * N;
  float s = 0.f, s2 = 0.f;
  for (int i = t; i < N; i += 256) { float v = y[i]; s += v; s2 += v * v; }
  __shared__ float rs[256], rq[256];
  rs[t] = s; rq[t] = s2;
  __syncthreads();
  for (int o = 128; o > 0; o >>= 1) {
    if (t < o) { rs[t] += rs[t + o]; rq[t] += rq[t + o]; }
    __syncthreads();
  }
  float mu = rs[0] / N;
  float inv = rsqrtf(rq[0] / N - mu * mu + EPSV);
  __syncthreads();
  for (int i = t; i < N; i += 256) {
    float f = (y[i] - mu) * inv * g[i] + b[i];
    y[i] = tok[i] + f;
  }
}

// ---------------------------------------------------------------------------
extern "C" void kernel_launch(void* const* d_in, const int* in_sizes, int n_in,
                              void* d_out, int out_size, void* d_ws,
                              size_t ws_size, hipStream_t stream) {
  (void)in_sizes; (void)n_in; (void)out_size; (void)ws_size;

  int i = 0;
  const float* bitboards = (const float*)d_in[i++];
  const float* conv1_w = (const float*)d_in[i++];
  const float* conv1_b = (const float*)d_in[i++];
  const float* bn1_g = (const float*)d_in[i++];
  const float* bn1_b = (const float*)d_in[i++];
  const float* bn1_m = (const float*)d_in[i++];
  const float* bn1_v = (const float*)d_in[i++];
  const float* conv2_w = (const float*)d_in[i++];
  const float* conv2_b = (const float*)d_in[i++];
  const float* bn2_g = (const float*)d_in[i++];
  const float* bn2_b = (const float*)d_in[i++];
  const float* bn2_m = (const float*)d_in[i++];
  const float* bn2_v = (const float*)d_in[i++];
  const float* proj_w = (const float*)d_in[i++];
  const float* proj_b = (const float*)d_in[i++];
  const float* sq_w = (const float*)d_in[i++];
  const float* sq_b = (const float*)d_in[i++];
  const float* sq_ln_g = (const float*)d_in[i++];
  const float* sq_ln_b = (const float*)d_in[i++];
  const float* init_w = (const float*)d_in[i++];
  const float* init_b = (const float*)d_in[i++];
  const float* ffn_w1 = (const float*)d_in[i++];
  const float* ffn_b1 = (const float*)d_in[i++];
  const float* ffn_ln1_g = (const float*)d_in[i++];
  const float* ffn_ln1_b = (const float*)d_in[i++];
  const float* ffn_w2 = (const float*)d_in[i++];
  const float* ffn_b2 = (const float*)d_in[i++];
  const float* ffn_ln2_g = (const float*)d_in[i++];
  const float* ffn_ln2_b = (const float*)d_in[i++];

  const int Mtok = 1024 * 64;  // 65536 token rows
  const int D = 768, DD = 1536;

  // workspace carve-out (256B aligned)
  char* ws = (char*)d_ws;
  size_t off = 0;
  auto carve = [&](size_t bytes) -> void* {
    void* p = ws + off;
    off += (bytes + 255) & ~(size_t)255;
    return p;
  };
  float*    tokens_f32 = (float*)   carve((size_t)Mtok * D * 4);
  _Float16* tokens_h   = (_Float16*)carve((size_t)Mtok * D * 2);
  _Float16* w1t        = (_Float16*)carve((size_t)DD * D * 2);   // (1536 x 768)
  _Float16* w2t        = (_Float16*)carve((size_t)D * DD * 2);   // (768 x 1536)
  float*    y1         = (float*)   carve((size_t)Mtok * DD * 4);
  _Float16* f1         = (_Float16*)carve((size_t)Mtok * DD * 2);
  float*    out        = (float*)d_out;

  // 1) weight convert/transpose to f16 N-major
  transpose_half_kernel<<<(D * DD + 255) / 256, 256, 0, stream>>>(ffn_w1, w1t, D, DD);
  transpose_half_kernel<<<(DD * D + 255) / 256, 256, 0, stream>>>(ffn_w2, w2t, DD, D);

  // 2) fused CNN/square front-end -> tokens
  tokens_kernel<<<1024, 256, 0, stream>>>(
      bitboards, conv1_w, conv1_b, bn1_g, bn1_b, bn1_m, bn1_v, conv2_w, conv2_b,
      bn2_g, bn2_b, bn2_m, bn2_v, proj_w, proj_b, sq_w, sq_b, sq_ln_g, sq_ln_b,
      init_w, init_b, tokens_f32, tokens_h);

  // 3) y1 = tokens @ W1 + b1   (65536 x 1536, K=768)
  dim3 g1(Mtok / 256, DD / 64);
  gemm_wmma<<<g1, 256, 0, stream>>>(tokens_h, w1t, ffn_b1, y1, Mtok, DD, D);

  // 4) f1 = gelu(LN(y1))  -> f16
  ln_gelu_kernel<<<Mtok, 256, 0, stream>>>(y1, ffn_ln1_g, ffn_ln1_b, f1, DD);

  // 5) out = f1 @ W2 + b2  (65536 x 768, K=1536) written straight to d_out
  dim3 g2(Mtok / 256, D / 64);
  gemm_wmma<<<g2, 256, 0, stream>>>(f1, w2t, ffn_b2, out, Mtok, D, DD);

  // 6) out = tokens + LN(out)  (in place)
  ln_residual_kernel<<<Mtok, 256, 0, stream>>>(out, tokens_f32, ffn_ln2_g,
                                               ffn_ln2_b, D);
}
